// AFNONet_35493609734881
// MI455X (gfx1250) — matrix operand order repaired and checked by reading, not compiled, not run on hardware
//
#include <hip/hip_runtime.h>
#include <stdint.h>
#include <stddef.h>

// ---------------- types ----------------
typedef unsigned short u16;
typedef float  v8f   __attribute__((ext_vector_type(8)));
typedef __bf16 v16bf __attribute__((ext_vector_type(16)));

union FragBF { unsigned int u[8]; v16bf v; };
union Pack8  { uint4 q; u16 s[8]; };

// ---------------- problem constants ----------------
#define B_     2
#define HP_    45
#define WP_    90
#define NPTOK  4050            // HP_*WP_
#define TT     8100            // B_*NPTOK tokens
#define EMB    768
#define HIDN   3072
#define DEPTH_ 8
#define NB_    16
#define BSZ    48
#define KEPT   23              // kept W-frequencies (out of 46)
#define MFREQ  (B_*HP_*KEPT)   // 2070 frequency "tokens"
#define NFE    (KEPT*EMB)      // 17664
#define SPEC   ((size_t)MFREQ*EMB)   // spectral buffer element count

// ---------------- GEMM tiling ----------------
#define TM 128
#define TN 64
#define TK 64
#define SA 72                  // LDS row stride (elems): 144B = 36 banks -> conflict-free
#define SB 72

__device__ __forceinline__ u16 f2bf(float x) {
  unsigned int u = __float_as_uint(x);
  unsigned int r = (u + 0x7FFFu + ((u >> 16) & 1u)) >> 16;  // RNE
  return (u16)r;
}

__device__ __forceinline__ float actf(float v, int act) {
  if (act == 1) v = 0.5f * v * (1.0f + erff(v * 0.70710678118654752f)); // exact GELU
  else if (act == 2) v = v > 0.f ? v : 0.f;                             // ReLU
  else if (act == 3) {                                                  // softshrink(0.01)
    float a = fabsf(v) - 0.01f;
    v = a > 0.f ? copysignf(a, v) : 0.f;
  }
  return v;
}

// =====================================================================
// Generic batched bf16 WMMA GEMM:  C = alpha*(A x B) [+ C] + bias, act
// A: (M x K) bf16 row-major, lda;  batch stride sAb (elems)
// B: (K x N) bf16 row-major, ldb;  batch stride sBb
// C: (M x N) f32  row-major, ldc;  batch stride sCb; optional bf16 mirror
// =====================================================================
__launch_bounds__(256)
__global__ void gemm_bf16_k(const u16* __restrict__ A, long long sAb, int lda,
                            const u16* __restrict__ B, long long sBb, int ldb,
                            float* __restrict__ C, long long sCb, int ldc,
                            u16* __restrict__ Cbf,
                            const float* __restrict__ bias, long long sBias,
                            int M, int N, int K,
                            float alpha, int accum, int act) {
  __shared__ __align__(16) u16 As[TM * SA];
  __shared__ __align__(16) u16 Bs[TN * SB];

  const int bz = blockIdx.z;
  A += (long long)bz * sAb;
  B += (long long)bz * sBb;
  C += (long long)bz * sCb;
  if (Cbf) Cbf += (long long)bz * sCb;
  const float* biasp = bias ? (bias + (long long)bz * sBias) : nullptr;

  const int m0   = blockIdx.y * TM;
  const int n0   = blockIdx.x * TN;
  const int tid  = threadIdx.x;
  const int lane = tid & 31;
  const int wave = tid >> 5;
  const int wm   = wave >> 1;      // 0..3  -> M sub-tile of 32
  const int wn   = wave & 1;       // 0..1  -> N sub-tile of 32
  const int hl   = lane >> 4;      // half-wave
  const int l16  = lane & 15;

  // hoisted staging predicates
  const int  ar   = tid >> 1;              // A row within tile
  const int  akc  = (tid & 1) * 32;        // A k-segment within tile
  const bool aRow = (m0 + ar) < M;
  const int  arC  = aRow ? (m0 + ar) : 0;  // clamped A row (always valid)
  const u16  aMsk = (u16)(0u - (unsigned)(aRow ? 1 : 0));
  const bool aAlg = ((lda & 7) == 0) && (((size_t)A & 15) == 0);
  const bool bAlg = ((n0 + TN) <= N) && ((ldb & 7) == 0) && (((size_t)B & 15) == 0);

  v8f zero = {0.f,0.f,0.f,0.f,0.f,0.f,0.f,0.f};
  v8f acc[2][2];
  acc[0][0] = zero; acc[0][1] = zero; acc[1][0] = zero; acc[1][1] = zero;

  for (int k0 = 0; k0 < K; k0 += TK) {
    // ---- stage A tile (TM x TK): thread -> row ar, 32 elems at akc
    {
      u16* dst = &As[ar * SA + akc];
      const u16* srcRow = A + (long long)arC * lda;
      int rem = K - (k0 + akc);
      if (aRow && aAlg && rem >= 32) {
        const u16* src = srcRow + (k0 + akc);
        uint4 v0 = *((const uint4*)src);
        uint4 v1 = *((const uint4*)(src + 8));
        uint4 v2 = *((const uint4*)(src + 16));
        uint4 v3 = *((const uint4*)(src + 24));
        *((uint4*)dst)        = v0;
        *((uint4*)(dst + 8))  = v1;
        *((uint4*)(dst + 16)) = v2;
        *((uint4*)(dst + 24)) = v3;
        if (k0 + TK < K) __builtin_prefetch(src + TK, 0, 1);  // global_prefetch_b8
      } else {
        // straight-line clamped slow path: load always executes (result always
        // used via AND-mask) so no predicated-load exec juggling is emitted
        #pragma unroll 1
        for (int j = 0; j < 32; j++) {
          int kg = k0 + akc + j;
          int kc = (kg < K) ? kg : 0;                        // clamped, valid
          u16 mk = (u16)(aMsk & (u16)(0u - (unsigned)(kg < K ? 1 : 0)));
          dst[j] = (u16)(srcRow[kc] & mk);
        }
      }
    }
    // ---- stage B tile transposed into Bs[n][k]  (TN x TK)
    if (bAlg && (k0 + TK) <= K) {
      // vector path: each pass loads 8 contiguous N-elems for one K row
      #pragma unroll
      for (int pass = 0; pass < 2; pass++) {
        int k  = (tid >> 3) + pass * 32;
        int n8 = (tid & 7) * 8;
        Pack8 t;
        t.q = *((const uint4*)(B + (long long)(k0 + k) * ldb + (n0 + n8)));
        #pragma unroll
        for (int j = 0; j < 8; j++) Bs[(n8 + j) * SB + k] = t.s[j];
      }
    } else {
      int n   = tid & 63;
      int kb  = (tid >> 6) * 16;
      int gn  = n0 + n;
      bool nOk = gn < N;
      int gnc = nOk ? gn : 0;
      u16 nMsk = (u16)(0u - (unsigned)(nOk ? 1 : 0));
      u16* dst = &Bs[n * SB + kb];
      #pragma unroll 1
      for (int j = 0; j < 16; j++) {
        int gk  = k0 + kb + j;
        int gkc = (gk < K) ? gk : 0;                         // clamped, valid
        u16 mk = (u16)(nMsk & (u16)(0u - (unsigned)(gk < K ? 1 : 0)));
        dst[j] = (u16)(B[(long long)gkc * ldb + gnc] & mk);
      }
    }
    __syncthreads();

    // ---- fragment loads (contiguous k = 16*hl + s in A and B) + 8x WMMA per staged tile
    #pragma unroll
    for (int kk = 0; kk < TK; kk += 32) {
      auto loadA = [&](int rBase) -> FragBF {
        FragBF f;
        const uint4* p0 = (const uint4*)&As[(rBase + l16) * SA + kk + hl * 16];
        const uint4* p1 = (const uint4*)&As[(rBase + l16) * SA + kk + hl * 16 + 8];
        uint4 x0 = *p0, x1 = *p1;
        f.u[0]=x0.x; f.u[1]=x0.y; f.u[2]=x0.z; f.u[3]=x0.w;
        f.u[4]=x1.x; f.u[5]=x1.y; f.u[6]=x1.z; f.u[7]=x1.w;
        return f;
      };
      auto loadB = [&](int nBase) -> FragBF {
        FragBF f;
        const uint4* p = (const uint4*)&Bs[(nBase + l16) * SB + kk + hl * 16];
        uint4 x0 = p[0], x1 = p[1];
        f.u[0]=x0.x; f.u[1]=x0.y; f.u[2]=x0.z; f.u[3]=x0.w;
        f.u[4]=x1.x; f.u[5]=x1.y; f.u[6]=x1.z; f.u[7]=x1.w;
        return f;
      };

      FragBF a0 = loadA(wm * 32), a1 = loadA(wm * 32 + 16);
      FragBF b0 = loadB(wn * 32), b1 = loadB(wn * 32 + 16);

      acc[0][0] = __builtin_amdgcn_wmma_f32_16x16x32_bf16(false, a0.v, false, b0.v,
                                                          (short)0, acc[0][0], false, false);
      acc[0][1] = __builtin_amdgcn_wmma_f32_16x16x32_bf16(false, a0.v, false, b1.v,
                                                          (short)0, acc[0][1], false, false);
      acc[1][0] = __builtin_amdgcn_wmma_f32_16x16x32_bf16(false, a1.v, false, b0.v,
                                                          (short)0, acc[1][0], false, false);
      acc[1][1] = __builtin_amdgcn_wmma_f32_16x16x32_bf16(false, a1.v, false, b1.v,
                                                          (short)0, acc[1][1], false, false);
    }
    __syncthreads();
  }

  // ---- epilogue: C layout -> lane: N=l16 ; VGPR v: M = v + hl*8
  const bool fullTile = ((m0 + TM) <= M) && ((n0 + TN) <= N);
  if (fullTile) {
    #pragma unroll
    for (int i = 0; i < 2; i++) {
      int rbase = m0 + wm * 32 + i * 16 + hl * 8;
      #pragma unroll
      for (int j = 0; j < 2; j++) {
        int col = n0 + wn * 32 + j * 16 + l16;
        float bv = biasp ? biasp[col] : 0.f;
        #pragma unroll
        for (int v = 0; v < 8; v++) {
          long long idx = (long long)(rbase + v) * ldc + col;
          float val = alpha * acc[i][j][v];
          if (accum) val += C[idx];
          val += bv;
          val = actf(val, act);
          C[idx] = val;
          if (Cbf) Cbf[idx] = f2bf(val);
        }
      }
    }
  } else {
    #pragma unroll
    for (int i = 0; i < 2; i++) {
      int rbase = m0 + wm * 32 + i * 16 + hl * 8;
      #pragma unroll
      for (int j = 0; j < 2; j++) {
        int col = n0 + wn * 32 + j * 16 + l16;
        if (col < N) {
          float bv = biasp ? biasp[col] : 0.f;
          #pragma unroll
          for (int v = 0; v < 8; v++) {
            int row = rbase + v;
            if (row < M) {
              long long idx = (long long)row * ldc + col;
              float val = alpha * acc[i][j][v];
              if (accum) val += C[idx];
              val += bv;
              val = actf(val, act);
              C[idx] = val;
              if (Cbf) Cbf[idx] = f2bf(val);
            }
          }
        }
      }
    }
  }
}

// =====================================================================
// LayerNorm over last dim (E=768), writes f32 + bf16 mirror
// =====================================================================
__launch_bounds__(256)
__global__ void layernorm_k(const float* __restrict__ x,
                            const float* __restrict__ w, const float* __restrict__ b,
                            float* __restrict__ y, u16* __restrict__ ybf, int E) {
  __shared__ float red[256];
  long long t = blockIdx.x;
  const float* xr = x + t * E;
  float s = 0.f;
  for (int i = threadIdx.x; i < E; i += 256) s += xr[i];
  red[threadIdx.x] = s; __syncthreads();
  for (int o = 128; o > 0; o >>= 1) { if (threadIdx.x < o) red[threadIdx.x] += red[threadIdx.x + o]; __syncthreads(); }
  float mean = red[0] / E; __syncthreads();
  float vs = 0.f;
  for (int i = threadIdx.x; i < E; i += 256) { float d = xr[i] - mean; vs += d * d; }
  red[threadIdx.x] = vs; __syncthreads();
  for (int o = 128; o > 0; o >>= 1) { if (threadIdx.x < o) red[threadIdx.x] += red[threadIdx.x + o]; __syncthreads(); }
  float inv = rsqrtf(red[0] / E + 1e-6f);
  for (int i = threadIdx.x; i < E; i += 256) {
    float o = (xr[i] - mean) * inv * w[i] + b[i];
    y[t * E + i] = o;
    ybf[t * E + i] = f2bf(o);
  }
}

// =====================================================================
// misc elementwise kernels
// =====================================================================
__global__ void im2col_k(const float* __restrict__ x, u16* __restrict__ Xp) {
  long long idx = (long long)blockIdx.x * blockDim.x + threadIdx.x;
  if (idx >= (long long)TT * 512) return;
  int k = (int)(idx & 511);
  long long t = idx >> 9;
  int b  = (int)(t / NPTOK);
  int p  = (int)(t % NPTOK);
  int hp = p / WP_, wp = p % WP_;
  int c  = k >> 8, py = (k >> 4) & 15, px = k & 15;
  float v = x[(((long long)b * 2 + c) * 720 + (hp * 16 + py)) * 1440 + (wp * 16 + px)];
  Xp[idx] = f2bf(v);
}

__global__ void transpose_cvt_k(const float* __restrict__ src, u16* __restrict__ dst,
                                int R, int C) {
  long long idx = (long long)blockIdx.x * blockDim.x + threadIdx.x;
  if (idx >= (long long)R * C) return;
  int r = (int)(idx / C), c = (int)(idx % C);
  dst[(long long)c * R + r] = f2bf(src[idx]);
}

__global__ void cvt_bf16_k(const float* __restrict__ src, u16* __restrict__ dst, long long n) {
  long long idx = (long long)blockIdx.x * blockDim.x + threadIdx.x;
  if (idx < n) dst[idx] = f2bf(src[idx]);
}

__global__ void split_w_k(const float* __restrict__ src, u16* __restrict__ wr,
                          u16* __restrict__ wi, int n) {
  int idx = blockIdx.x * blockDim.x + threadIdx.x;
  if (idx >= n) return;
  wr[idx] = f2bf(src[2 * idx]);
  wi[idx] = f2bf(src[2 * idx + 1]);
}

__global__ void split_b_k(const float* __restrict__ src, float* __restrict__ br,
                          float* __restrict__ bi, int n) {
  int idx = blockIdx.x * blockDim.x + threadIdx.x;
  if (idx >= n) return;
  br[idx] = src[2 * idx];
  bi[idx] = src[2 * idx + 1];
}

__global__ void add_pos_k(float* __restrict__ h, const float* __restrict__ pos) {
  long long idx = (long long)blockIdx.x * blockDim.x + threadIdx.x;
  if (idx >= (long long)TT * EMB) return;
  long long t = idx / EMB;
  int e = (int)(idx % EMB);
  int p = (int)(t % NPTOK);
  h[idx] += pos[(long long)e * NPTOK + p];
}

__global__ void add3_k(const float* __restrict__ a, const float* __restrict__ b,
                       const float* __restrict__ c, float* __restrict__ o, long long n) {
  long long idx = (long long)blockIdx.x * blockDim.x + threadIdx.x;
  if (idx < n) o[idx] = a[idx] + b[idx] + c[idx];
}

__global__ void add2_k(const float* __restrict__ a, const float* __restrict__ b,
                       float* __restrict__ o, long long n) {
  long long idx = (long long)blockIdx.x * blockDim.x + threadIdx.x;
  if (idx < n) o[idx] = a[idx] + b[idx];
}

// DFT matrices (ortho norm folded in), stored bf16 for WMMA A-operands
__global__ void gen_dft_k(u16* Dr, u16* Di, u16* Hr, u16* Hi,
                          u16* HIr, u16* HIi, u16* Gr, u16* Gi) {
  int idx = blockIdx.x * blockDim.x + threadIdx.x;
  const float PI2  = 6.28318530717958647692f;
  const float is90 = 0.105409255338945986f;   // 1/sqrt(90)
  const float is45 = 0.149071198499985996f;   // 1/sqrt(45)
  if (idx < KEPT * WP_) {                                   // forward W-DFT (23x90)
    int f = idx / WP_, w = idx % WP_;
    float th = PI2 * (float)(f * w) / 90.f;
    Dr[idx] = f2bf(cosf(th) * is90);
    Di[idx] = f2bf(-sinf(th) * is90);
  } else if (idx < KEPT * WP_ + 45 * 45) {                  // H-DFT (45x45) fwd+inv
    int j = idx - KEPT * WP_;
    int a = j / 45, c = j % 45;
    float th = PI2 * (float)(a * c) / 45.f;
    float co = cosf(th) * is45, si = sinf(th) * is45;
    Hr[j]  = f2bf(co);  Hi[j]  = f2bf(-si);
    HIr[j] = f2bf(co);  HIi[j] = f2bf(si);
  } else if (idx < KEPT * WP_ + 45 * 45 + WP_ * KEPT) {     // inverse rDFT-W (90x23)
    int j = idx - KEPT * WP_ - 45 * 45;
    int w = j / KEPT, f = j % KEPT;
    float cf = (f == 0) ? 1.f : 2.f;                        // Hermitian fold
    float th = PI2 * (float)(f * w) / 90.f;
    Gr[j] = f2bf(cf * cosf(th) * is90);
    Gi[j] = f2bf(-cf * sinf(th) * is90);
  }
}

__global__ void unpatch_k(const float* __restrict__ Hout, float* __restrict__ out) {
  long long idx = (long long)blockIdx.x * blockDim.x + threadIdx.x;
  if (idx >= (long long)B_ * 2 * 720 * 1440) return;
  int xx = (int)(idx % 1440);
  int yy = (int)((idx / 1440) % 720);
  int oc = (int)((idx / (1440LL * 720)) % 2);
  int b  = (int)(idx / (1440LL * 720 * 2));
  int hp = yy >> 4, p = yy & 15, wp = xx >> 4, q = xx & 15;
  long long t = (long long)b * NPTOK + hp * WP_ + wp;
  int o = (p * 16 + q) * 2 + oc;
  out[idx] = Hout[t * 512 + o];
}

// =====================================================================
// host side
// =====================================================================
static inline void gemmL(hipStream_t s,
    const void* A, long long sA, int lda,
    const void* Bm, long long sB, int ldb,
    void* C, long long sC, int ldc, void* Cbf,
    const void* bias, long long sBias,
    int M, int N, int K, float alpha, int accum, int act, int batch) {
  dim3 g((N + TN - 1) / TN, (M + TM - 1) / TM, batch);
  gemm_bf16_k<<<g, 256, 0, s>>>((const u16*)A, sA, lda, (const u16*)Bm, sB, ldb,
                                (float*)C, sC, ldc, (u16*)Cbf,
                                (const float*)bias, sBias, M, N, K, alpha, accum, act);
}

static inline dim3 ew(long long n) { return dim3((unsigned)((n + 255) / 256)); }

extern "C" void kernel_launch(void* const* d_in, const int* in_sizes, int n_in,
                              void* d_out, int out_size, void* d_ws, size_t ws_size,
                              hipStream_t stream) {
  (void)in_sizes; (void)n_in; (void)out_size; (void)ws_size;

  const float* x       = (const float*)d_in[0];
  const float* patch_w = (const float*)d_in[1];
  const float* patch_b = (const float*)d_in[2];
  const float* pos     = (const float*)d_in[3];
  const float* n1w     = (const float*)d_in[4];
  const float* n1b     = (const float*)d_in[5];
  const float* w1      = (const float*)d_in[6];
  const float* b1      = (const float*)d_in[7];
  const float* w2      = (const float*)d_in[8];
  const float* b2      = (const float*)d_in[9];
  const float* n2w     = (const float*)d_in[10];
  const float* n2b     = (const float*)d_in[11];
  const float* fc1w    = (const float*)d_in[12];
  const float* fc1b    = (const float*)d_in[13];
  const float* fc2w    = (const float*)d_in[14];
  const float* fc2b    = (const float*)d_in[15];
  const float* headw   = (const float*)d_in[16];

  // ---- workspace bump allocator ----
  char* W = (char*)d_ws;
  size_t off = 0;
  auto AL = [&](size_t bytes) -> char* {
    off = (off + 255) & ~(size_t)255;
    char* p = W + off; off += bytes; return p;
  };
  const size_t TE = (size_t)TT * EMB;

  float* h     = (float*)AL(TE * 4);
  float* ln    = (float*)AL(TE * 4);
  float* y     = (float*)AL(TE * 4);
  float* yaf   = (float*)AL(TE * 4);
  u16*   lnbf  = (u16*)  AL(TE * 2);
  float* hid   = (float*)AL((size_t)TT * HIDN * 4);
  u16*   hidbf = (u16*)  AL((size_t)TT * HIDN * 2);
  u16*   Xp    = (u16*)  AL((size_t)TT * 512 * 2);
  u16*   pbt   = (u16*)  AL((size_t)512 * EMB * 2);
  u16*   hbt   = (u16*)  AL((size_t)EMB * 512 * 2);
  float* Hout  = (float*)AL((size_t)TT * 512 * 4);

  float *S1r = (float*)AL(SPEC * 4), *S1i = (float*)AL(SPEC * 4);
  float *Zr  = (float*)AL(SPEC * 4), *Zi  = (float*)AL(SPEC * 4);
  float *O1r = (float*)AL(SPEC * 4), *O1i = (float*)AL(SPEC * 4);
  float *O2r = (float*)AL(SPEC * 4), *O2i = (float*)AL(SPEC * 4);
  float *S2r = (float*)AL(SPEC * 4), *S2i = (float*)AL(SPEC * 4);
  u16 *S1rb = (u16*)AL(SPEC * 2), *S1ib = (u16*)AL(SPEC * 2);
  u16 *Zrb  = (u16*)AL(SPEC * 2), *Zib  = (u16*)AL(SPEC * 2);
  u16 *O1rb = (u16*)AL(SPEC * 2), *O1ib = (u16*)AL(SPEC * 2);
  u16 *O2rb = (u16*)AL(SPEC * 2), *O2ib = (u16*)AL(SPEC * 2);
  u16 *S2rb = (u16*)AL(SPEC * 2), *S2ib = (u16*)AL(SPEC * 2);

  u16* fc1wb = (u16*)AL((size_t)EMB * HIDN * 2);
  u16* fc2wb = (u16*)AL((size_t)HIDN * EMB * 2);
  u16 *w1r = (u16*)AL(NB_*BSZ*BSZ*2), *w1i = (u16*)AL(NB_*BSZ*BSZ*2);
  u16 *w2r = (u16*)AL(NB_*BSZ*BSZ*2), *w2i = (u16*)AL(NB_*BSZ*BSZ*2);
  float *b1r = (float*)AL(NB_*BSZ*4), *b1i = (float*)AL(NB_*BSZ*4);
  float *b2r = (float*)AL(NB_*BSZ*4), *b2i = (float*)AL(NB_*BSZ*4);

  u16 *Dr = (u16*)AL(KEPT*WP_*2), *Di = (u16*)AL(KEPT*WP_*2);
  u16 *Hr = (u16*)AL(45*45*2),    *Hi = (u16*)AL(45*45*2);
  u16 *HIr= (u16*)AL(45*45*2),    *HIi= (u16*)AL(45*45*2);
  u16 *Gr = (u16*)AL(WP_*KEPT*2), *Gi = (u16*)AL(WP_*KEPT*2);

  // ---- one-time prep ----
  im2col_k<<<ew((long long)TT * 512), 256, 0, stream>>>(x, Xp);
  transpose_cvt_k<<<ew((long long)EMB * 512), 256, 0, stream>>>(patch_w, pbt, EMB, 512);
  transpose_cvt_k<<<ew((long long)512 * EMB), 256, 0, stream>>>(headw,  hbt, 512, EMB);
  gen_dft_k<<<ew(KEPT*WP_ + 45*45 + WP_*KEPT), 256, 0, stream>>>(Dr, Di, Hr, Hi, HIr, HIi, Gr, Gi);

  // ---- patch embed: h = Xp(T,512) x pbt(512,768) + patch_b ; + pos ----
  gemmL(stream, Xp, 0, 512, pbt, 0, EMB, h, 0, EMB, nullptr,
        patch_b, 0, TT, EMB, 512, 1.f, 0, 0, 1);
  add_pos_k<<<ew((long long)TE), 256, 0, stream>>>(h, pos);

  // ---- layers ----
  for (int d = 0; d < DEPTH_; d++) {
    const float* n1w_d = n1w + (size_t)d * EMB;
    const float* n1b_d = n1b + (size_t)d * EMB;
    const float* n2w_d = n2w + (size_t)d * EMB;
    const float* n2b_d = n2b + (size_t)d * EMB;
    const float* w1_d  = w1  + (size_t)d * NB_ * BSZ * BSZ * 2;
    const float* w2_d  = w2  + (size_t)d * NB_ * BSZ * BSZ * 2;
    const float* b1_d  = b1  + (size_t)d * NB_ * BSZ * 2;
    const float* b2_d  = b2  + (size_t)d * NB_ * BSZ * 2;
    const float* f1w_d = fc1w + (size_t)d * EMB * HIDN;
    const float* f1b_d = fc1b + (size_t)d * HIDN;
    const float* f2w_d = fc2w + (size_t)d * HIDN * EMB;
    const float* f2b_d = fc2b + (size_t)d * EMB;

    cvt_bf16_k<<<ew((long long)EMB * HIDN), 256, 0, stream>>>(f1w_d, fc1wb, (long long)EMB * HIDN);
    cvt_bf16_k<<<ew((long long)HIDN * EMB), 256, 0, stream>>>(f2w_d, fc2wb, (long long)HIDN * EMB);
    split_w_k<<<ew(NB_*BSZ*BSZ), 256, 0, stream>>>(w1_d, w1r, w1i, NB_*BSZ*BSZ);
    split_w_k<<<ew(NB_*BSZ*BSZ), 256, 0, stream>>>(w2_d, w2r, w2i, NB_*BSZ*BSZ);
    split_b_k<<<ew(NB_*BSZ), 256, 0, stream>>>(b1_d, b1r, b1i, NB_*BSZ);
    split_b_k<<<ew(NB_*BSZ), 256, 0, stream>>>(b2_d, b2r, b2i, NB_*BSZ);

    // LN1
    layernorm_k<<<dim3(TT), 256, 0, stream>>>(h, n1w_d, n1b_d, ln, lnbf, EMB);

    // forward W-DFT (batch over B*Hp=90 slabs):  S1 = D x Y   (M=23,K=90,N=768)
    gemmL(stream, Dr, 0, WP_, lnbf, (long long)WP_*EMB, EMB,
          S1r, (long long)KEPT*EMB, EMB, S1rb, nullptr, 0,
          KEPT, EMB, WP_, 1.f, 0, 0, B_*HP_);
    gemmL(stream, Di, 0, WP_, lnbf, (long long)WP_*EMB, EMB,
          S1i, (long long)KEPT*EMB, EMB, S1ib, nullptr, 0,
          KEPT, EMB, WP_, 1.f, 0, 0, B_*HP_);

    // forward H-DFT (batch over B, complex):  Z = H x S1  (M=45,K=45,N=23*768)
    gemmL(stream, Hr, 0, 45, S1rb, (long long)HP_*NFE, NFE, Zr, (long long)HP_*NFE, NFE,
          nullptr, nullptr, 0, 45, NFE, 45, 1.f, 0, 0, B_);
    gemmL(stream, Hi, 0, 45, S1ib, (long long)HP_*NFE, NFE, Zr, (long long)HP_*NFE, NFE,
          Zrb, nullptr, 0, 45, NFE, 45, -1.f, 1, 0, B_);
    gemmL(stream, Hr, 0, 45, S1ib, (long long)HP_*NFE, NFE, Zi, (long long)HP_*NFE, NFE,
          nullptr, nullptr, 0, 45, NFE, 45, 1.f, 0, 0, B_);
    gemmL(stream, Hi, 0, 45, S1rb, (long long)HP_*NFE, NFE, Zi, (long long)HP_*NFE, NFE,
          Zib, nullptr, 0, 45, NFE, 45, 1.f, 1, 0, B_);

    // block-diag complex MLP (batch over 16 blocks; M=2070,K=N=48)
    // layer 1: O1 = relu(Z * W1 + b1)
    gemmL(stream, Zrb, 48, EMB, w1r, (long long)BSZ*BSZ, BSZ, O1r, 48, EMB,
          nullptr, nullptr, 0, MFREQ, BSZ, BSZ, 1.f, 0, 0, NB_);
    gemmL(stream, Zib, 48, EMB, w1i, (long long)BSZ*BSZ, BSZ, O1r, 48, EMB,
          O1rb, b1r, BSZ, MFREQ, BSZ, BSZ, -1.f, 1, 2, NB_);
    gemmL(stream, Zib, 48, EMB, w1r, (long long)BSZ*BSZ, BSZ, O1i, 48, EMB,
          nullptr, nullptr, 0, MFREQ, BSZ, BSZ, 1.f, 0, 0, NB_);
    gemmL(stream, Zrb, 48, EMB, w1i, (long long)BSZ*BSZ, BSZ, O1i, 48, EMB,
          O1ib, b1i, BSZ, MFREQ, BSZ, BSZ, 1.f, 1, 2, NB_);
    // layer 2: O2 = softshrink(O1 * W2 + b2)
    gemmL(stream, O1rb, 48, EMB, w2r, (long long)BSZ*BSZ, BSZ, O2r, 48, EMB,
          nullptr, nullptr, 0, MFREQ, BSZ, BSZ, 1.f, 0, 0, NB_);
    gemmL(stream, O1ib, 48, EMB, w2i, (long long)BSZ*BSZ, BSZ, O2r, 48, EMB,
          O2rb, b2r, BSZ, MFREQ, BSZ, BSZ, -1.f, 1, 3, NB_);
    gemmL(stream, O1ib, 48, EMB, w2r, (long long)BSZ*BSZ, BSZ, O2i, 48, EMB,
          nullptr, nullptr, 0, MFREQ, BSZ, BSZ, 1.f, 0, 0, NB_);
    gemmL(stream, O1rb, 48, EMB, w2i, (long long)BSZ*BSZ, BSZ, O2i, 48, EMB,
          O2ib, b2i, BSZ, MFREQ, BSZ, BSZ, 1.f, 1, 3, NB_);

    // inverse H-DFT (complex): S2 = HI x O2
    gemmL(stream, HIr, 0, 45, O2rb, (long long)HP_*NFE, NFE, S2r, (long long)HP_*NFE, NFE,
          nullptr, nullptr, 0, 45, NFE, 45, 1.f, 0, 0, B_);
    gemmL(stream, HIi, 0, 45, O2ib, (long long)HP_*NFE, NFE, S2r, (long long)HP_*NFE, NFE,
          S2rb, nullptr, 0, 45, NFE, 45, -1.f, 1, 0, B_);
    gemmL(stream, HIr, 0, 45, O2ib, (long long)HP_*NFE, NFE, S2i, (long long)HP_*NFE, NFE,
          nullptr, nullptr, 0, 45, NFE, 45, 1.f, 0, 0, B_);
    gemmL(stream, HIi, 0, 45, O2rb, (long long)HP_*NFE, NFE, S2i, (long long)HP_*NFE, NFE,
          S2ib, nullptr, 0, 45, NFE, 45, 1.f, 1, 0, B_);

    // inverse rDFT-W (batch 90 slabs):  yaf = Gr x S2r + Gi x S2i (M=90,K=23,N=768)
    gemmL(stream, Gr, 0, KEPT, S2rb, (long long)KEPT*EMB, EMB,
          yaf, (long long)WP_*EMB, EMB, nullptr, nullptr, 0,
          WP_, EMB, KEPT, 1.f, 0, 0, B_*HP_);
    gemmL(stream, Gi, 0, KEPT, S2ib, (long long)KEPT*EMB, EMB,
          yaf, (long long)WP_*EMB, EMB, nullptr, nullptr, 0,
          WP_, EMB, KEPT, 1.f, 1, 0, B_*HP_);

    // y = irfft + LN1(h) + h   (AFNO inner residual + outer residual)
    add3_k<<<ew((long long)TE), 256, 0, stream>>>(yaf, ln, h, y, (long long)TE);

    // LN2 + MLP
    layernorm_k<<<dim3(TT), 256, 0, stream>>>(y, n2w_d, n2b_d, ln, lnbf, EMB);
    gemmL(stream, lnbf, 0, EMB, fc1wb, 0, HIDN, hid, 0, HIDN, hidbf,
          f1b_d, 0, TT, HIDN, EMB, 1.f, 0, 1, 1);                  // GELU
    gemmL(stream, hidbf, 0, HIDN, fc2wb, 0, EMB, ln, 0, EMB, nullptr,
          f2b_d, 0, TT, EMB, HIDN, 1.f, 0, 0, 1);
    add2_k<<<ew((long long)TE), 256, 0, stream>>>(ln, y, h, (long long)TE);
  }

  // ---- head + unpatchify ----
  cvt_bf16_k<<<ew((long long)TE), 256, 0, stream>>>(h, lnbf, (long long)TE);
  gemmL(stream, lnbf, 0, EMB, hbt, 0, 512, Hout, 0, 512, nullptr,
        nullptr, 0, TT, 512, EMB, 1.f, 0, 0, 1);
  unpatch_k<<<ew((long long)B_ * 2 * 720 * 1440), 256, 0, stream>>>(Hout, (float*)d_out);
}
